// GruDecoder_5669356834995
// MI455X (gfx1250) — compile-verified, matrix operand order checked by reading, not run on hardware
//
#include <hip/hip_runtime.h>
#include <hip/hip_bf16.h>
#include <stdint.h>

// ---------------- Types ----------------
typedef __bf16 bf16;
typedef __attribute__((ext_vector_type(16))) __bf16 v16bf;
typedef __attribute__((ext_vector_type(8)))  float  v8f;

struct Frag32 { uint4 lo, hi; };   // 32 bytes == one v16bf fragment
static_assert(sizeof(Frag32) == 32, "frag size");

#define HID   1024
#define ODIM  768
#define TM    128
#define TN    128
#define TK    32
#define LSTR  40        // LDS row stride in bf16 (80 B, 16B-aligned, staggers banks)

// CDNA5 async global->LDS copy (ASYNCcnt path). GVS addressing:
//   global addr = SADDR + VADDR.u32 + offset ; LDS addr = VDST.u32 + offset
__device__ __forceinline__ void async_copy_b128(uint32_t lds_off, const void* sbase,
                                                uint32_t voff) {
    asm volatile("global_load_async_to_lds_b128 %0, %1, %2"
                 :: "v"(lds_off), "v"(voff), "s"(sbase) : "memory");
}
__device__ __forceinline__ void async_copy_b128_o16(uint32_t lds_off, const void* sbase,
                                                    uint32_t voff) {
    asm volatile("global_load_async_to_lds_b128 %0, %1, %2 offset:16"
                 :: "v"(lds_off), "v"(voff), "s"(sbase) : "memory");
}
__device__ __forceinline__ void wait_async0() {
    asm volatile("s_wait_asynccnt 0x0" ::: "memory");
}

// LDS byte offset of a __shared__ object: low 32 bits of its generic address
__device__ __forceinline__ uint32_t lds_off_of(const void* p) {
    return (uint32_t)(uintptr_t)p;
}

// ---------------- f32 -> bf16 convert ----------------
__global__ void cvt_f32_bf16(const float* __restrict__ s, bf16* __restrict__ d, int n) {
    int i = blockIdx.x * blockDim.x + threadIdx.x;
    if (i < n) d[i] = (bf16)s[i];
}

// ---------------- WMMA GEMM: C[M,N] = A[M,K] * W[N,K]^T (+bias) ----------------
// grid = (N/TN, M/TM), block = 256 threads = 8 wave32s (4 row-blocks x 2 col-blocks)
__global__ __launch_bounds__(256) void gemm_bf16_wmma(
    const bf16* __restrict__ A,      // [M,K] row-major, bf16
    const bf16* __restrict__ W,      // [N,K] row-major, bf16  (B-matrix = W^T)
    const float* __restrict__ bias,  // [N] or nullptr
    float* __restrict__ C,           // [M,N] f32
    bf16*  __restrict__ Cbf,         // optional bf16 mirror of C, or nullptr
    int M, int N, int K)
{
    __shared__ bf16 As[2][TM][LSTR];
    __shared__ bf16 Bs[2][TN][LSTR];

    const int t      = threadIdx.x;
    const int blockN = blockIdx.x * TN;
    const int blockM = blockIdx.y * TM;

    const int lane  = t & 31;
    const int wave  = t >> 5;        // 0..7
    const int wr    = wave & 3;      // 32-row block within tile
    const int wc    = wave >> 2;     // 64-col block within tile
    const int lrow  = lane & 15;     // M (A) / N (B) index within 16
    const int lhalf = lane >> 4;     // K half select per ISA 16-bit layout

    // cooperative async tile loader: 256 threads, each moves 32 B per tile
    const int lr = t & 127;          // tile row
    const int lh = t >> 7;           // which 16-col half of the 32-wide K chunk
    const uint32_t voff = (uint32_t)((lr * K + lh * 16) * sizeof(bf16)); // loop-invariant
    uint32_t ldsA[2], ldsB[2];
    ldsA[0] = lds_off_of(&As[0][lr][lh * 16]);
    ldsA[1] = lds_off_of(&As[1][lr][lh * 16]);
    ldsB[0] = lds_off_of(&Bs[0][lr][lh * 16]);
    ldsB[1] = lds_off_of(&Bs[1][lr][lh * 16]);

    v8f acc[2][4];
#pragma unroll
    for (int mt = 0; mt < 2; ++mt)
#pragma unroll
        for (int nt = 0; nt < 4; ++nt) acc[mt][nt] = (v8f){0,0,0,0,0,0,0,0};

    auto load_tiles = [&](int k0, int buf) {
        const char* sa = (const char*)(A + (size_t)blockM * K + k0);   // uniform -> SGPRs
        const char* sb = (const char*)(W + (size_t)blockN * K + k0);
        async_copy_b128    (ldsA[buf], sa, voff);
        async_copy_b128_o16(ldsA[buf], sa, voff);   // offset:16 shifts LDS and global together
        async_copy_b128    (ldsB[buf], sb, voff);
        async_copy_b128_o16(ldsB[buf], sb, voff);
    };

    load_tiles(0, 0);
    wait_async0();
    __syncthreads();

    int buf = 0;
    for (int k0 = 0; k0 < K; k0 += TK, buf ^= 1) {
        if (k0 + TK < K) load_tiles(k0 + TK, buf ^ 1);

        // A fragments: 16x32 bf16 per ISA layout (lanes 0-15: K 0-7 & 16-23; lanes 16-31: K 8-15 & 24-31)
        v16bf af[2];
#pragma unroll
        for (int mt = 0; mt < 2; ++mt) {
            const bf16* p = &As[buf][wr * 32 + mt * 16 + lrow][lhalf * 8];
            Frag32 f;
            f.lo = *(const uint4*)p;          // K = lhalf*8 .. +7
            f.hi = *(const uint4*)(p + 16);   // K = 16+lhalf*8 .. +7
            af[mt] = __builtin_bit_cast(v16bf, f);
        }
        // B fragments (K x 16 column per lane; W rows are B columns -> contiguous in LDS)
        v16bf bfr[4];
#pragma unroll
        for (int nt = 0; nt < 4; ++nt) {
            const bf16* p = &Bs[buf][wc * 64 + nt * 16 + lrow][lhalf * 8];
            Frag32 f;
            f.lo = *(const uint4*)p;
            f.hi = *(const uint4*)(p + 16);
            bfr[nt] = __builtin_bit_cast(v16bf, f);
        }

#pragma unroll
        for (int mt = 0; mt < 2; ++mt)
#pragma unroll
            for (int nt = 0; nt < 4; ++nt)
                acc[mt][nt] = __builtin_amdgcn_wmma_f32_16x16x32_bf16(
                    false, af[mt], false, bfr[nt], (short)0, acc[mt][nt], false, false);

        // next-buffer async fills must have landed in LDS before anyone reads them
        wait_async0();
        __syncthreads();
    }

    // store: D layout — VGPR r holds row (r + 8*lhalf), lane&15 is column
#pragma unroll
    for (int mt = 0; mt < 2; ++mt) {
        const int rbase = blockM + wr * 32 + mt * 16 + lhalf * 8;
#pragma unroll
        for (int nt = 0; nt < 4; ++nt) {
            const int gcol = blockN + wc * 64 + nt * 16 + lrow;
            const float bv = bias ? bias[gcol] : 0.0f;
#pragma unroll
            for (int r = 0; r < 8; ++r) {
                const float v = acc[mt][nt][r] + bv;
                const size_t off = (size_t)(rbase + r) * N + gcol;
                C[off] = v;
                if (Cbf) Cbf[off] = (bf16)v;
            }
        }
    }
}

// ---------------- GRU gate elementwise ----------------
// gi/gh: [B, 3H] (r,z,n gate order); h: [B,H] f32 in/out; h_bf mirrors h in bf16
__global__ __launch_bounds__(256) void gru_gate(
    const float* __restrict__ gi, const float* __restrict__ gh,
    const float* __restrict__ b_ih, const float* __restrict__ b_hh,
    float* __restrict__ h, bf16* __restrict__ h_bf)
{
    const int idx = blockIdx.x * blockDim.x + threadIdx.x;   // B*H threads
    const int b = idx >> 10;            // H == 1024
    const int j = idx & (HID - 1);
    const size_t base = (size_t)b * 3 * HID;

    const float ir = gi[base + j]           + b_ih[j];
    const float iz = gi[base + HID + j]     + b_ih[HID + j];
    const float in = gi[base + 2 * HID + j] + b_ih[2 * HID + j];
    const float hr = gh[base + j]           + b_hh[j];
    const float hz = gh[base + HID + j]     + b_hh[HID + j];
    const float hn = gh[base + 2 * HID + j] + b_hh[2 * HID + j];

    const float r = 1.0f / (1.0f + __expf(-(ir + hr)));
    const float z = 1.0f / (1.0f + __expf(-(iz + hz)));
    const float n = tanhf(in + r * hn);

    const float hv   = h[idx];
    const float hnew = (1.0f - z) * n + z * hv;
    h[idx]    = hnew;
    h_bf[idx] = (bf16)hnew;
}

// ---------------- Host orchestration ----------------
extern "C" void kernel_launch(void* const* d_in, const int* in_sizes, int n_in,
                              void* d_out, int out_size, void* d_ws, size_t ws_size,
                              hipStream_t stream) {
    const float* src    = (const float*)d_in[0];   // [S,B,O]   (only src[0] used)
    const float* hidden = (const float*)d_in[2];   // [1,B,H]
    const float* W_ih   = (const float*)d_in[3];   // [3H,O]
    const float* W_hh   = (const float*)d_in[4];   // [3H,H]
    const float* b_ih   = (const float*)d_in[5];
    const float* b_hh   = (const float*)d_in[6];
    const float* W_fc   = (const float*)d_in[7];   // [O,H]
    const float* b_fc   = (const float*)d_in[8];
    float* out = (float*)d_out;                    // [T,B,O]

    const int H = HID, O = ODIM;
    const int B = in_sizes[2] / H;                 // 256
    const int T = in_sizes[1] / B;                 // tgt is [T,B]

    // workspace carve-out
    char* wsp = (char*)d_ws;
    auto carve = [&](size_t bytes) {
        char* p = wsp;
        wsp += (bytes + 255) & ~(size_t)255;
        return (void*)p;
    };
    bf16*  W_ih_bf = (bf16*)carve((size_t)3 * H * O * sizeof(bf16));
    bf16*  W_hh_bf = (bf16*)carve((size_t)3 * H * H * sizeof(bf16));
    bf16*  W_fc_bf = (bf16*)carve((size_t)O * H * sizeof(bf16));
    bf16*  x_bf    = (bf16*)carve((size_t)B * O * sizeof(bf16));
    bf16*  h_bf    = (bf16*)carve((size_t)B * H * sizeof(bf16));
    float* h_f32   = (float*)carve((size_t)B * H * sizeof(float));
    float* gi      = (float*)carve((size_t)B * 3 * H * sizeof(float));
    float* gh      = (float*)carve((size_t)B * 3 * H * sizeof(float));

    auto cvt = [&](const float* s, bf16* d, int n) {
        cvt_f32_bf16<<<(n + 255) / 256, 256, 0, stream>>>(s, d, n);
    };
    cvt(W_ih, W_ih_bf, 3 * H * O);
    cvt(W_hh, W_hh_bf, 3 * H * H);
    cvt(W_fc, W_fc_bf, O * H);
    cvt(src,    x_bf, B * O);     // x0 = src[0]
    cvt(hidden, h_bf, B * H);     // h0
    hipMemcpyAsync(h_f32, hidden, (size_t)B * H * sizeof(float),
                   hipMemcpyDeviceToDevice, stream);

    const dim3 blk(256);
    const dim3 gGates(3 * H / TN, B / TM);   // 24 x 2
    const dim3 gFc(O / TN, B / TM);          // 6 x 2

    for (int t = 0; t < T; ++t) {
        // gi = x @ W_ih^T ; gh = h @ W_hh^T  (biases folded in gate kernel)
        gemm_bf16_wmma<<<gGates, blk, 0, stream>>>(x_bf, W_ih_bf, nullptr, gi, nullptr, B, 3 * H, O);
        gemm_bf16_wmma<<<gGates, blk, 0, stream>>>(h_bf, W_hh_bf, nullptr, gh, nullptr, B, 3 * H, H);
        // GRU cell update (writes h_f32 + h_bf)
        gru_gate<<<(B * H) / 256, blk, 0, stream>>>(gi, gh, b_ih, b_hh, h_f32, h_bf);
        // y = h_new @ W_fc^T + b_fc  -> d_out[t], bf16 mirror becomes next x
        gemm_bf16_wmma<<<gFc, blk, 0, stream>>>(h_bf, W_fc_bf, b_fc,
                                                out + (size_t)t * B * O, x_bf, B, O, H);
    }
    (void)n_in; (void)out_size; (void)ws_size;
}
